// MemTransformerLM_15891378995471
// MI455X (gfx1250) — matrix operand order repaired and checked by reading, not compiled, not run on hardware
//
#include <hip/hip_runtime.h>
#include <hip/hip_bf16.h>

// ---------------------------------------------------------------------------
// CDNA5 (gfx1250) WMMA types & helpers
// ---------------------------------------------------------------------------
typedef __attribute__((ext_vector_type(16))) __bf16 v16bf;
typedef __attribute__((ext_vector_type(8)))  __bf16 v8bf;
typedef __attribute__((ext_vector_type(8)))  float  v8f;
typedef __attribute__((ext_vector_type(4)))  unsigned int u32x4;
typedef __attribute__((ext_vector_type(4)))  int i32x4;
typedef __attribute__((ext_vector_type(8)))  int i32x8;

static __device__ __forceinline__ unsigned short f32_to_bf16_rne(float f) {
  union { float f; unsigned int u; } c; c.f = f;
  unsigned int u = c.u;
  u += 0x7FFFu + ((u >> 16) & 1u);
  return (unsigned short)(u >> 16);
}
static __device__ __forceinline__ __bf16 us_as_bf(unsigned short u) {
  union { unsigned short u; __bf16 b; } c; c.u = u; return c.b;
}
static __device__ __forceinline__ v16bf cat8(v8bf lo, v8bf hi) {
  v16bf r;
#pragma unroll
  for (int i = 0; i < 8; ++i) { r[i] = lo[i]; r[i + 8] = hi[i]; }
  return r;
}

// A-matrix fragment (16x32 bf16), per ISA: lane m holds K = {half*8+0..7} and
// {16+half*8+0..7}; p points at row start + kbase (kbase multiple of 32).
static __device__ __forceinline__ v16bf load_frag_A(const unsigned short* p, int half) {
  v8bf lo = *(const v8bf*)(p + half * 8);
  v8bf hi = *(const v8bf*)(p + 16 + half * 8);
  return cat8(lo, hi);
}
// B-matrix fragment (32x16 bf16): lane holds column n and K = {half*8+0..7} and
// {16+half*8+0..7} (same order as A). Memory is B^T row-major (row = n).
static __device__ __forceinline__ v16bf load_frag_Bt(const unsigned short* p, int half) {
  v8bf lo = *(const v8bf*)(p + half * 8);
  v8bf hi = *(const v8bf*)(p + 16 + half * 8);
  return cat8(lo, hi);
}

static __device__ __forceinline__ v8f wmma_bf16(v16bf a, v16bf b, v8f c) {
  // D = A(16x32) * B(32x16) + C, f32 accumulate
  return __builtin_amdgcn_wmma_f32_16x16x32_bf16(false, a, false, b, (short)0, c,
                                                 false, false);
}

// Async copy: 16 bytes global -> LDS, tracked by ASYNCcnt.
static __device__ __forceinline__ void async_g2l_b128(unsigned int lds_byte_addr,
                                                      const void* gptr) {
  unsigned long long g = (unsigned long long)(uintptr_t)gptr;
  asm volatile("global_load_async_to_lds_b128 %0, %1, off"
               :: "v"(lds_byte_addr), "v"(g) : "memory");
}
static __device__ __forceinline__ void wait_asynccnt0() {
  asm volatile("s_wait_asynccnt 0x0" ::: "memory");
}

// Tensor Data Mover: DMA a 2D bf16 tile (rows x cols, row stride = cols) from
// global memory into LDS. Builds the D# descriptor per CDNA5 ISA:
//   group0: [1:0]=count=1, [63:32]=lds_addr, [120:64]=global_addr, [127:126]=2
//   group1: [17:16]=data_size(1=2B), dims/strides as annotated below.
// Tracked by TENSORcnt; wait with s_wait_tensorcnt.
static __device__ __forceinline__ void tdm_load_2d_bf16(unsigned int lds_byte_addr,
                                                        const void* gptr,
                                                        unsigned int rows,
                                                        unsigned int cols) {
  unsigned long long ga = (unsigned long long)(uintptr_t)gptr;
  u32x4 g0;
  g0[0] = 1u;                                         // count=1 (valid, user mode)
  g0[1] = lds_byte_addr;                              // lds_addr
  g0[2] = (unsigned int)(ga & 0xFFFFFFFFu);           // global_addr[31:0]
  g0[3] = (unsigned int)((ga >> 32) & 0x01FFFFFFu)    // global_addr[56:32]
          | 0x80000000u;                              // type=2 ("image")
  i32x8 g1;
  g1[0] = 0x10000;                                    // data_size=1 (2 bytes)
  g1[1] = (int)(cols << 16);                          // tensor_dim0[15:0] @ 63:48
  g1[2] = (int)((cols >> 16) | (rows << 16));         // td0[31:16] | td1[15:0]
  g1[3] = (int)((rows >> 16) | (cols << 16));         // td1[31:16] | tile_dim0
  g1[4] = (int)rows;                                  // tile_dim1 (tile_dim2=0)
  g1[5] = (int)cols;                                  // tensor_dim0_stride[31:0]
  g1[6] = 0;                                          // stride hi / dim1_stride
  g1[7] = 0;
  i32x4 z4 = {0, 0, 0, 0};
#if defined(__clang_major__) && (__clang_major__ >= 23)
  i32x8 z8 = {0, 0, 0, 0, 0, 0, 0, 0};
  __builtin_amdgcn_tensor_load_to_lds(g0, g1, z4, z4, z8, 0);
#else
  __builtin_amdgcn_tensor_load_to_lds(g0, g1, z4, z4, 0);
#endif
}

// ---------------------------------------------------------------------------
// Kernel 0: fp32 -> bf16 conversion
// ---------------------------------------------------------------------------
__global__ void k_f32_to_bf16(const float* __restrict__ src,
                              unsigned short* __restrict__ dst, int n) {
  int i = blockIdx.x * blockDim.x + threadIdx.x;
  if (i < n) dst[i] = f32_to_bf16_rne(src[i]);
}

// ---------------------------------------------------------------------------
// Kernel 1: QKV GEMM [16384,1024]x[1024,4096] with fused phi epilogue.
// Block = 8 waves; wave computes 16 rows x 64 cols (one head-quarter chunk,
// so the phi row-normalization over dh=64 stays inside the wave).
// ---------------------------------------------------------------------------
__global__ __launch_bounds__(256) void k_qkv_phi(
    const unsigned short* __restrict__ A,   // h   bf16 [16384,1024]
    const unsigned short* __restrict__ Bw,  // Wqkv bf16 [4096,1024] (= B^T)
    unsigned short* __restrict__ qkvphi)    // [(b*16+h)*4+s][t*64+d] bf16
{
  const int lane = threadIdx.x & 31;
  const int wave = threadIdx.x >> 5;
  const int half = lane >> 4;
  const int ll   = lane & 15;
  const int m0   = blockIdx.x * 16;
  const int colbase = blockIdx.y * 512 + wave * 64;

  v8f acc[4] = {};
  const unsigned short* arow = A + (unsigned long long)(m0 + ll) * 1024ull;
#pragma unroll 1
  for (int kk = 0; kk < 1024; kk += 32) {
    __builtin_prefetch(arow + kk + 64, 0, 1);  // global_prefetch_b8
    v16bf af = load_frag_A(arow + kk, half);
#pragma unroll
    for (int j = 0; j < 4; ++j) {
      const unsigned short* brow =
          Bw + (unsigned long long)(colbase + j * 16 + ll) * 1024ull + kk;
      v16bf bf = load_frag_Bt(brow, half);
      acc[j] = wmma_bf16(af, bf, acc[j]);
    }
  }

  const int s    = (colbase >> 6) & 3;  // 0:q 1:k1 2:k2 3:v
  const int head = colbase >> 8;

  float vals[4][8];
#pragma unroll
  for (int j = 0; j < 4; ++j)
#pragma unroll
    for (int r = 0; r < 8; ++r) vals[j][r] = acc[j][r];

  if (s < 3) {  // phi = (elu(x)+1) / rowsum
#pragma unroll
    for (int j = 0; j < 4; ++j)
#pragma unroll
      for (int r = 0; r < 8; ++r) {
        float x = vals[j][r];
        vals[j][r] = (x > 0.0f) ? (x + 1.0f) : __expf(x);
      }
#pragma unroll
    for (int r = 0; r < 8; ++r) {
      float rs = vals[0][r] + vals[1][r] + vals[2][r] + vals[3][r];
      rs += __shfl_xor(rs, 1, 32);
      rs += __shfl_xor(rs, 2, 32);
      rs += __shfl_xor(rs, 4, 32);
      rs += __shfl_xor(rs, 8, 32);  // sum within 16-lane half (matches C layout)
      float inv = 1.0f / rs;
      vals[0][r] *= inv; vals[1][r] *= inv; vals[2][r] *= inv; vals[3][r] *= inv;
    }
  }

#pragma unroll
  for (int r = 0; r < 8; ++r) {
    int m = m0 + r + 8 * half;  // row = l*B + b
    int t = m >> 6;             // l (B = 64)
    int b = m & 63;             // b
    unsigned long long base =
        (unsigned long long)((b * 16 + head) * 4 + s) * 16384ull;
#pragma unroll
    for (int j = 0; j < 4; ++j) {
      int d = j * 16 + ll;
      qkvphi[base + (unsigned long long)t * 64ull + d] = f32_to_bf16_rne(vals[j][r]);
    }
  }
}

// ---------------------------------------------------------------------------
// Kernel 2: causal linear attention == fast-weight scan with W0=0.
// One workgroup per (b,h); 16 waves, wave w owns query rows [16w,16w+16).
//  - K1, K2 (256x64 bf16 tiles) are DMA'd into LDS by the Tensor Data Mover
//    (TENSOR_LOAD_TO_LDS, TENSORcnt), issued once by wave 0.
//  - V is staged TRANSPOSED into LDS (coalesced b128 reads, 16-bit scatter),
//    so every O-GEMM B-fragment is two contiguous ds_load_b128.
//  - pi0 row staged via GLOBAL_LOAD_ASYNC_TO_LDS (ASYNCcnt).
// S = tril(Q K^T) staged through LDS (bf16) to reshape C->A layout.
// Dynamic LDS: 64KB S + 32KB K1 + 32KB K2 + 32KB Vt + 1KB pi = 161KB.
// ---------------------------------------------------------------------------
__global__ __launch_bounds__(512) void k_fastweight(
    const unsigned short* __restrict__ qkvphi,
    const float* __restrict__ pi0,             // [16,256]
    unsigned short* __restrict__ layer_out)    // bf16 [16384,1024]
{
  extern __shared__ unsigned short smem[];
  // ushort-index layout:
  //   [     0, 32768)  S staging: 2 streams x 16 waves x 1024
  //   [ 32768, 49152)  K1 [256][64]          (byte  65536, TDM)
  //   [ 49152, 65536)  K2 [256][64]          (byte  98304, TDM)
  //   [ 65536, 81920)  Vt [64][256]          (byte 131072, transposed)
  //   [ 81920, 82432)  pi row (256 floats)   (byte 163840, async)
  const int lane = threadIdx.x & 31;
  const int wave = threadIdx.x >> 5;  // 0..15
  const int half = lane >> 4;
  const int ll   = lane & 15;
  const int bh   = blockIdx.x;        // b*16 + h
  const int b    = bh >> 4;
  const int hh   = bh & 15;

  const unsigned short* Qp  = qkvphi + (unsigned long long)(bh * 4 + 0) * 16384ull;
  const unsigned short* K1p = qkvphi + (unsigned long long)(bh * 4 + 1) * 16384ull;
  const unsigned short* K2p = qkvphi + (unsigned long long)(bh * 4 + 2) * 16384ull;
  const unsigned short* Vp  = qkvphi + (unsigned long long)(bh * 4 + 3) * 16384ull;

  const unsigned int ldsBase =
      (unsigned int)(unsigned long long)(uintptr_t)(void*)smem;

  // ---- TDM: DMA K1 and K2 tiles into LDS (one wave issues, TENSORcnt) ----
  if (wave == 0) {
    tdm_load_2d_bf16(ldsBase + 65536u, K1p, 256u, 64u);
    tdm_load_2d_bf16(ldsBase + 98304u, K2p, 256u, 64u);
  }
  // ---- async: stage this head's pi row (1KB) into LDS (ASYNCcnt) ----
  if (threadIdx.x < 64) {
    async_g2l_b128(ldsBase + 163840u + (unsigned)threadIdx.x * 16u,
                   (const char*)(pi0 + hh * 256) + threadIdx.x * 16);
  }
  // ---- stage V transposed: Vt[d][s], coalesced reads, 16-bit LDS scatter ----
#pragma unroll 1
  for (int i = threadIdx.x; i < 2048; i += 512) {  // 2048 x 8-elem chunks
    int s = i >> 3;
    int dbase = (i & 7) * 8;
    const uint4 vv = *(const uint4*)(Vp + s * 64 + dbase);
    unsigned int w4[4] = {vv.x, vv.y, vv.z, vv.w};
#pragma unroll
    for (int e2 = 0; e2 < 4; ++e2) {
      smem[65536 + (dbase + 2 * e2 + 0) * 256 + s] = (unsigned short)(w4[e2] & 0xFFFFu);
      smem[65536 + (dbase + 2 * e2 + 1) * 256 + s] = (unsigned short)(w4[e2] >> 16);
    }
  }

  // overlap: load this wave's Q fragments from global while DMA runs
  const int trow = wave * 16 + ll;  // this lane's query row for A fragments
  v16bf qf0 = load_frag_A(Qp + trow * 64 + 0,  half);
  v16bf qf1 = load_frag_A(Qp + trow * 64 + 32, half);

  wait_asynccnt0();
  __builtin_amdgcn_s_wait_tensorcnt(0);
  __syncthreads();

  const unsigned short* K1l = smem + 32768;
  const unsigned short* K2l = smem + 49152;
  const unsigned short* Vt  = smem + 65536;
  const float* piL = (const float*)(smem + 81920);
  unsigned short* sS0 = smem + wave * 1024;
  unsigned short* sS1 = smem + 16384 + wave * 1024;

  v8f o1[4] = {}; v8f o2[4] = {};
  const int nchunks = (wave >> 2) + 1;  // causal: only key chunks <= diagonal
#pragma unroll 1
  for (int kc = 0; kc < nchunks; ++kc) {
#pragma unroll
    for (int j = 0; j < 4; ++j) {
      int srow = kc * 64 + j * 16 + ll;  // key index for B fragments
      v8f s1 = {};
      s1 = wmma_bf16(qf0, load_frag_Bt(K1l + srow * 64 + 0,  half), s1);
      s1 = wmma_bf16(qf1, load_frag_Bt(K1l + srow * 64 + 32, half), s1);
      v8f s2 = {};
      s2 = wmma_bf16(qf0, load_frag_Bt(K2l + srow * 64 + 0,  half), s2);
      s2 = wmma_bf16(qf1, load_frag_Bt(K2l + srow * 64 + 32, half), s2);
#pragma unroll
      for (int r = 0; r < 8; ++r) {        // causal mask + bf16 + LDS stage
        int t  = wave * 16 + r + 8 * half; // query index of this C element
        int sg = kc * 64 + j * 16 + ll;    // key index
        float x1 = (sg <= t) ? s1[r] : 0.0f;
        float x2 = (sg <= t) ? s2[r] : 0.0f;
        sS0[(r + 8 * half) * 64 + j * 16 + ll] = f32_to_bf16_rne(x1);
        sS1[(r + 8 * half) * 64 + j * 16 + ll] = f32_to_bf16_rne(x2);
      }
    }
    // O += S @ V  (DS ops are in-order per wave; per-wave LDS region)
#pragma unroll
    for (int ks = 0; ks < 2; ++ks) {
      v16bf a1 = load_frag_A(sS0 + ll * 64 + ks * 32, half);
      v16bf a2 = load_frag_A(sS1 + ll * 64 + ks * 32, half);
#pragma unroll
      for (int jd = 0; jd < 4; ++jd) {
        int d = jd * 16 + ll;
        v16bf vb = load_frag_Bt(Vt + (unsigned)d * 256u + kc * 64 + ks * 32, half);
        o1[jd] = wmma_bf16(a1, vb, o1[jd]);
        o2[jd] = wmma_bf16(a2, vb, o2[jd]);
      }
    }
  }

  // gate with clipped pi, scale = 1/sqrt(dh) = 0.125, store bf16 [l*B+b, h*64+d]
#pragma unroll
  for (int r = 0; r < 8; ++r) {
    int t = wave * 16 + r + 8 * half;
    float pi = piL[t];
    pi = fminf(fmaxf(pi, 0.0f), 1.0f);
#pragma unroll
    for (int jd = 0; jd < 4; ++jd) {
      int d = jd * 16 + ll;
      float v = 0.125f * (pi * o1[jd][r] + (1.0f - pi) * o2[jd][r]);
      layer_out[(unsigned long long)(t * 64 + b) * 1024ull + hh * 64 + d] =
          f32_to_bf16_rne(v);
    }
  }
}

// ---------------------------------------------------------------------------
// Kernel 3: output GEMM [16384,1024]x[1024,1024] + residual + LayerNorm fused.
// Block = 8 waves covering 16 rows x full 1024 cols so LN reduces in LDS.
// ---------------------------------------------------------------------------
__global__ __launch_bounds__(256) void k_out_ln(
    const unsigned short* __restrict__ A,   // layer_out bf16 [16384,1024]
    const unsigned short* __restrict__ Bw,  // W_o bf16 [1024,1024] (= B^T)
    const float* __restrict__ hres,         // residual fp32 [16384,1024]
    const float* __restrict__ gamma,
    const float* __restrict__ beta,
    float* __restrict__ out)
{
  __shared__ float red[8][16][2];
  __shared__ float mu_s[16];
  __shared__ float rstd_s[16];
  const int lane = threadIdx.x & 31;
  const int wave = threadIdx.x >> 5;  // 0..7
  const int half = lane >> 4;
  const int ll   = lane & 15;
  const int m0   = blockIdx.x * 16;
  const int colbase = wave * 128;

  v8f acc[8] = {};
  const unsigned short* arow = A + (unsigned long long)(m0 + ll) * 1024ull;
#pragma unroll 1
  for (int kk = 0; kk < 1024; kk += 32) {
    __builtin_prefetch(arow + kk + 64, 0, 1);
    v16bf af = load_frag_A(arow + kk, half);
#pragma unroll
    for (int j = 0; j < 8; ++j) {
      v16bf bf = load_frag_Bt(
          Bw + (unsigned long long)(colbase + j * 16 + ll) * 1024ull + kk, half);
      acc[j] = wmma_bf16(af, bf, acc[j]);
    }
  }

  // residual add
  float x[8][8];
#pragma unroll
  for (int j = 0; j < 8; ++j)
#pragma unroll
    for (int r = 0; r < 8; ++r) {
      int m = m0 + r + 8 * half;
      int d = colbase + j * 16 + ll;
      x[j][r] = acc[j][r] + hres[(unsigned long long)m * 1024ull + d];
    }

  // per-row partial sums within wave (halves reduce independently)
#pragma unroll
  for (int r = 0; r < 8; ++r) {
    float s = 0.f, ss = 0.f;
#pragma unroll
    for (int j = 0; j < 8; ++j) { s += x[j][r]; ss += x[j][r] * x[j][r]; }
    s  += __shfl_xor(s, 1, 32);  s  += __shfl_xor(s, 2, 32);
    s  += __shfl_xor(s, 4, 32);  s  += __shfl_xor(s, 8, 32);
    ss += __shfl_xor(ss, 1, 32); ss += __shfl_xor(ss, 2, 32);
    ss += __shfl_xor(ss, 4, 32); ss += __shfl_xor(ss, 8, 32);
    if (ll == 0) {  // lane 0 -> rows 0-7, lane 16 -> rows 8-15
      red[wave][r + 8 * half][0] = s;
      red[wave][r + 8 * half][1] = ss;
    }
  }
  __syncthreads();
  if (threadIdx.x < 16) {
    float s = 0.f, ss = 0.f;
    for (int wv = 0; wv < 8; ++wv) { s += red[wv][threadIdx.x][0]; ss += red[wv][threadIdx.x][1]; }
    float mu  = s * (1.0f / 1024.0f);
    float var = ss * (1.0f / 1024.0f) - mu * mu;
    mu_s[threadIdx.x]   = mu;
    rstd_s[threadIdx.x] = rsqrtf(var + 1e-5f);
  }
  __syncthreads();

#pragma unroll
  for (int j = 0; j < 8; ++j) {
    int d = colbase + j * 16 + ll;
    float g = gamma[d], bt = beta[d];
#pragma unroll
    for (int r = 0; r < 8; ++r) {
      int mrow = r + 8 * half;
      float v = (x[j][r] - mu_s[mrow]) * rstd_s[mrow] * g + bt;
      out[(unsigned long long)(m0 + mrow) * 1024ull + d] = v;
    }
  }
}

// ---------------------------------------------------------------------------
// Launch
// ---------------------------------------------------------------------------
extern "C" void kernel_launch(void* const* d_in, const int* in_sizes, int n_in,
                              void* d_out, int out_size, void* d_ws, size_t ws_size,
                              hipStream_t stream) {
  const float* h     = (const float*)d_in[0];  // [256,64,1024]
  const float* W_qkv = (const float*)d_in[1];  // [4096,1024]
  const float* W_o   = (const float*)d_in[2];  // [1024,1024]
  const float* gamma = (const float*)d_in[3];  // [1024]
  const float* beta  = (const float*)d_in[4];  // [1024]
  const float* pi0   = (const float*)d_in[5];  // [16,256]
  float* out = (float*)d_out;

  char* ws = (char*)d_ws;
  unsigned short* hA   = (unsigned short*)(ws);                       // 32 MB
  unsigned short* Wq   = (unsigned short*)(ws + (33554432ull));       //  8 MB
  unsigned short* Wo   = (unsigned short*)(ws + (41943040ull));       //  2 MB
  unsigned short* qkv  = (unsigned short*)(ws + (44040192ull));       // 128 MB
  unsigned short* lout = (unsigned short*)(ws + (178257920ull));      // 32 MB

  // allow 161KB dynamic LDS for the attention kernel (WGP has 320KB)
  (void)hipFuncSetAttribute((const void*)k_fastweight,
                            hipFuncAttributeMaxDynamicSharedMemorySize, 164864);

  k_f32_to_bf16<<<(16384 * 1024 + 255) / 256, 256, 0, stream>>>(h, hA, 16384 * 1024);
  k_f32_to_bf16<<<(4096 * 1024 + 255) / 256, 256, 0, stream>>>(W_qkv, Wq, 4096 * 1024);
  k_f32_to_bf16<<<(1024 * 1024 + 255) / 256, 256, 0, stream>>>(W_o, Wo, 1024 * 1024);

  // QKV projection + phi:   grid = (16384/16 M-tiles, 4096/512 N-tiles)
  k_qkv_phi<<<dim3(1024, 8), 256, 0, stream>>>(hA, Wq, qkv);
  // causal linear attention, one workgroup per (b,h), 161KB dynamic LDS
  k_fastweight<<<1024, 512, 164864, stream>>>(qkv, pi0, lout);
  // output projection + residual + LayerNorm
  k_out_ln<<<1024, 256, 0, stream>>>(lout, Wo, h, gamma, beta, out);
}
